// MambaSequenceClassifier_77352361000982
// MI455X (gfx1250) — compile-verified, matrix operand order checked by reading, not run on hardware
//
#include <hip/hip_runtime.h>
#include <hip/hip_bf16.h>
#include <math.h>

// ---------------- model constants ----------------
#define BATCH    32
#define SEQL     1000
#define ROWS     (BATCH * SEQL)       // 32000
#define DM       128                  // d_model
#define DI       256                  // d_inner
#define DSTATE   16
#define DTR      8                    // dt_rank
#define NXDBL    (DTR + 2 * DSTATE)   // 40
#define NLAYERS  6
#define NCLS     71
#define INCH     12

typedef __attribute__((ext_vector_type(16))) _Float16 v16h;
typedef __attribute__((ext_vector_type(8)))  _Float16 h8;
typedef __attribute__((ext_vector_type(8)))  float    v8f;

union V16 { v16h v; h8 h[2]; };

// ---------------- helpers ----------------
__device__ __forceinline__ float silu_f(float s) {
    return s / (1.0f + __expf(-s));
}

__device__ __forceinline__ float block_reduce_128(float v, float* red, int t) {
    red[t] = v; __syncthreads();
    #pragma unroll
    for (int s = 64; s > 0; s >>= 1) {
        if (t < s) red[t] += red[t + s];
        __syncthreads();
    }
    float r = red[0]; __syncthreads();
    return r;
}

// ---------------- kernels ----------------

// f32 -> f16 elementwise (weight conversion)
__global__ void k_f32_to_f16(const float* __restrict__ in, _Float16* __restrict__ out, int n) {
    int i = blockIdx.x * blockDim.x + threadIdx.x;
    if (i < n) out[i] = (_Float16)in[i];
}

// input projection: h[row, d] = sum_c x[row,c] * inp_w[d,c] + inp_b[d]; one block per row, 128 threads
__global__ void k_input_proj(const float* __restrict__ x, const float* __restrict__ inp_w,
                             const float* __restrict__ inp_b, float* __restrict__ hst) {
    __shared__ float xr[INCH];
    int row = blockIdx.x, t = threadIdx.x;
    if (t < INCH) xr[t] = x[(size_t)row * INCH + t];
    __syncthreads();
    float s = inp_b[t];
    #pragma unroll
    for (int c = 0; c < INCH; ++c) s += xr[c] * inp_w[t * INCH + c];
    hst[(size_t)row * DM + t] = s;
}

// layernorm over DM=128, f16 output; one block per row, 128 threads
__global__ void k_layernorm_f16(const float* __restrict__ hst, const float* __restrict__ w,
                                const float* __restrict__ b, _Float16* __restrict__ out) {
    __shared__ float red[DM];
    int row = blockIdx.x, t = threadIdx.x;
    float v = hst[(size_t)row * DM + t];
    float mu = block_reduce_128(v, red, t) * (1.0f / DM);
    float d = v - mu;
    float var = block_reduce_128(d * d, red, t) * (1.0f / DM);
    out[(size_t)row * DM + t] = (_Float16)(d * rsqrtf(var + 1e-5f) * w[t] + b[t]);
}

// WMMA GEMM: out[M,N] = A[M,K](f16) * W[N,K]^T(f16) (+ resid); f32 accumulate.
// One 16x16 output tile per wave; K stepped by 32 with v_wmma_f32_16x16x32_f16.
// Fragment layout per CDNA5 ISA 7.12.2: lane r(0..15) row/col r, hi half selects K blocks
// {hi*8..hi*8+7} and {16+hi*8..16+hi*8+7} -> two contiguous b128 loads per fragment.
__global__ void k_gemm_wmma(const _Float16* __restrict__ A, const _Float16* __restrict__ W,
                            const float* __restrict__ resid, float* __restrict__ out,
                            int M, int N, int K) {
    int wave = threadIdx.x >> 5;
    int lane = threadIdx.x & 31;
    int r    = lane & 15;
    int hi   = lane >> 4;
    int n0 = (blockIdx.x * (blockDim.x >> 5) + wave) * 16;
    int m0 = blockIdx.y * 16;
    if (n0 >= N || m0 >= M) return;   // wave-uniform: EXEC stays all-ones for active waves

    const _Float16* arow = A + (size_t)(m0 + r) * K;
    const _Float16* brow = W + (size_t)(n0 + r) * K;

    v8f acc = {};
    for (int k = 0; k < K; k += 32) {
        V16 a, b;
        a.h[0] = *(const h8*)(arow + k + hi * 8);
        a.h[1] = *(const h8*)(arow + k + 16 + hi * 8);
        b.h[0] = *(const h8*)(brow + k + hi * 8);
        b.h[1] = *(const h8*)(brow + k + 16 + hi * 8);
        acc = __builtin_amdgcn_wmma_f32_16x16x32_f16(false, a.v, false, b.v,
                                                     (short)0, acc, false, false);
    }
    // D layout: element v -> (m0 + v + 8*hi, n0 + r)
    #pragma unroll
    for (int v = 0; v < 8; ++v) {
        size_t idx = (size_t)(m0 + v + 8 * hi) * N + (n0 + r);
        float o = acc[v];
        if (resid) o += resid[idx];
        out[idx] = o;
    }
}

// causal depthwise conv (K=4) + SiLU: reads u-part of xz (cols 0..255, row stride 512)
__global__ void k_conv_silu(const float* __restrict__ xz, const float* __restrict__ cw,
                            const float* __restrict__ cb, float* __restrict__ u) {
    int idx = blockIdx.x * blockDim.x + threadIdx.x;
    if (idx >= ROWS * DI) return;
    int c = idx & (DI - 1);
    int row = idx >> 8;
    int l = row % SEQL;
    int base = row - l;                 // b*SEQL
    const float* w = cw + c * 4;
    float s = cb[c];
    #pragma unroll
    for (int k = 0; k < 4; ++k) {
        int ls = l - 3 + k;
        if (ls >= 0) s += w[k] * xz[(size_t)(base + ls) * (2 * DI) + c];
    }
    u[idx] = silu_f(s);
}

// x_dbl[row, j] = sum_c u[row,c] * xp_w[j,c], j < 40; one block (64 threads) per row
__global__ void k_xdbl(const float* __restrict__ u, const float* __restrict__ xp_w,
                       float* __restrict__ xdbl) {
    __shared__ float ur[DI];
    int row = blockIdx.x, t = threadIdx.x;
    for (int i = t; i < DI; i += 64) ur[i] = u[(size_t)row * DI + i];
    __syncthreads();
    if (t < NXDBL) {
        const float* wr = xp_w + t * DI;
        float s = 0.0f;
        for (int c = 0; c < DI; ++c) s += ur[c] * wr[c];
        xdbl[(size_t)row * NXDBL + t] = s;
    }
}

// dt[row, d] = softplus(sum_r xdbl[row,r] * dtp_w[d,r] + dtp_b[d]); one block (256) per row
__global__ void k_dtproj(const float* __restrict__ xdbl, const float* __restrict__ dtp_w,
                         const float* __restrict__ dtp_b, float* __restrict__ dt) {
    __shared__ float dr[DTR];
    int row = blockIdx.x, t = threadIdx.x;
    if (t < DTR) dr[t] = xdbl[(size_t)row * NXDBL + t];
    __syncthreads();
    float s = dtp_b[t];
    #pragma unroll
    for (int r = 0; r < DTR; ++r) s += dr[r] * dtp_w[t * DTR + r];
    dt[(size_t)row * DI + t] = (s > 20.0f) ? s : log1pf(expf(s));
}

// selective scan: one block per batch, 256 threads (one per inner channel),
// 16 states per thread in registers, B/C broadcast through LDS per timestep.
// Writes y into the u-columns of xz (cols 0..255).
__global__ void k_scan(const float* __restrict__ dt, const float* __restrict__ xdbl,
                       const float* __restrict__ u, const float* __restrict__ A_log,
                       float* __restrict__ xz_y) {
    __shared__ float Bs[DSTATE], Cs[DSTATE];
    int b = blockIdx.x, d = threadIdx.x;
    float A[DSTATE], hst[DSTATE];
    #pragma unroll
    for (int s = 0; s < DSTATE; ++s) {
        A[s] = -__expf(A_log[d * DSTATE + s]);
        hst[s] = 0.0f;
    }
    for (int l = 0; l < SEQL; ++l) {
        size_t row = (size_t)b * SEQL + l;
        if (d < 2 * DSTATE) {
            const float* xd = xdbl + row * NXDBL + DTR;   // [B(16) | C(16)]
            if (d < DSTATE) Bs[d] = xd[d];
            else            Cs[d - DSTATE] = xd[d];
        }
        __syncthreads();
        float dtv = dt[row * DI + d];
        float uv  = u[row * DI + d];
        float p   = dtv * uv;
        float y   = 0.0f;
        #pragma unroll
        for (int s = 0; s < DSTATE; ++s) {
            float dA = __expf(dtv * A[s]);
            hst[s] = dA * hst[s] + p * Bs[s];
            y += hst[s] * Cs[s];
        }
        xz_y[row * (2 * DI) + d] = y;
        __syncthreads();
    }
}

// ymod[row,d] = (y + u*Dp[d]) * silu(z) as f16 (feeds out_proj WMMA)
__global__ void k_ymod(const float* __restrict__ xz, const float* __restrict__ u,
                       const float* __restrict__ Dp, _Float16* __restrict__ ymod) {
    int idx = blockIdx.x * blockDim.x + threadIdx.x;
    if (idx >= ROWS * DI) return;
    int c = idx & (DI - 1);
    int row = idx >> 8;
    float y = xz[(size_t)row * (2 * DI) + c];
    float z = xz[(size_t)row * (2 * DI) + DI + c];
    float v = (y + u[idx] * Dp[c]) * silu_f(z);
    ymod[idx] = (_Float16)v;
}

// final layernorm + mean-pool over L: one block per batch, 128 threads
__global__ void k_final_ln_pool(const float* __restrict__ hst, const float* __restrict__ w,
                                const float* __restrict__ b, float* __restrict__ pooled) {
    __shared__ float red[DM];
    int bb = blockIdx.x, t = threadIdx.x;
    float acc = 0.0f;
    for (int l = 0; l < SEQL; ++l) {
        float v = hst[((size_t)bb * SEQL + l) * DM + t];
        float mu = block_reduce_128(v, red, t) * (1.0f / DM);
        float d = v - mu;
        float var = block_reduce_128(d * d, red, t) * (1.0f / DM);
        acc += d * rsqrtf(var + 1e-5f) * w[t] + b[t];
    }
    pooled[bb * DM + t] = acc * (1.0f / SEQL);
}

// head: out[b, cls] = pooled[b,:] . head_w[cls,:] + head_b[cls]
__global__ void k_head(const float* __restrict__ pooled, const float* __restrict__ head_w,
                       const float* __restrict__ head_b, float* __restrict__ out) {
    __shared__ float p[DM];
    int b = blockIdx.x, t = threadIdx.x;
    p[t] = pooled[b * DM + t];
    __syncthreads();
    if (t < NCLS) {
        float s = head_b[t];
        const float* wr = head_w + t * DM;
        for (int d = 0; d < DM; ++d) s += p[d] * wr[d];
        out[b * NCLS + t] = s;
    }
}

// ---------------- host ----------------
extern "C" void kernel_launch(void* const* d_in, const int* in_sizes, int n_in,
                              void* d_out, int out_size, void* d_ws, size_t ws_size,
                              hipStream_t stream) {
    const float* x      = (const float*)d_in[0];
    const float* inp_w  = (const float*)d_in[1];
    const float* inp_b  = (const float*)d_in[2];
    const float* ln_w   = (const float*)d_in[3];
    const float* ln_b   = (const float*)d_in[4];
    const float* in_w   = (const float*)d_in[5];
    const float* conv_w = (const float*)d_in[6];
    const float* conv_b = (const float*)d_in[7];
    const float* xp_w   = (const float*)d_in[8];
    const float* dtp_w  = (const float*)d_in[9];
    const float* dtp_b  = (const float*)d_in[10];
    const float* A_log  = (const float*)d_in[11];
    const float* Dp     = (const float*)d_in[12];
    const float* out_w  = (const float*)d_in[13];
    const float* fn_w   = (const float*)d_in[14];
    const float* fn_b   = (const float*)d_in[15];
    const float* head_w = (const float*)d_in[16];
    const float* head_b = (const float*)d_in[17];
    float* out = (float*)d_out;

    char* ws = (char*)d_ws;
    size_t off = 0;
    auto alloc = [&](size_t bytes) {
        void* p = ws + off;
        off = (off + bytes + 255) & ~(size_t)255;
        return p;
    };
    float*    hbuf   = (float*)   alloc((size_t)ROWS * DM * 4);        // residual stream
    _Float16* normh  = (_Float16*)alloc((size_t)ROWS * DM * 2);        // LN output f16
    float*    xz     = (float*)   alloc((size_t)ROWS * 2 * DI * 4);    // in_proj out (u|z); y overwrites u-cols
    float*    ubuf   = (float*)   alloc((size_t)ROWS * DI * 4);        // conv+silu
    float*    xdbl   = (float*)   alloc((size_t)ROWS * NXDBL * 4);
    float*    dtbuf  = (float*)   alloc((size_t)ROWS * DI * 4);
    _Float16* ymod   = (_Float16*)alloc((size_t)ROWS * DI * 2);
    _Float16* inw16  = (_Float16*)alloc((size_t)NLAYERS * 2 * DI * DM * 2);
    _Float16* outw16 = (_Float16*)alloc((size_t)NLAYERS * DM * DI * 2);
    float*    pooled = (float*)   alloc((size_t)BATCH * DM * 4);

    // one-time weight conversion to f16
    {
        int n1 = NLAYERS * 2 * DI * DM;   // 393216
        int n2 = NLAYERS * DM * DI;       // 196608
        k_f32_to_f16<<<(n1 + 255) / 256, 256, 0, stream>>>(in_w, inw16, n1);
        k_f32_to_f16<<<(n2 + 255) / 256, 256, 0, stream>>>(out_w, outw16, n2);
    }

    // input projection
    k_input_proj<<<ROWS, DM, 0, stream>>>(x, inp_w, inp_b, hbuf);

    const int elemBlocks = (ROWS * DI + 255) / 256;

    for (int i = 0; i < NLAYERS; ++i) {
        const float*    lnw  = ln_w   + i * DM;
        const float*    lnb  = ln_b   + i * DM;
        const _Float16* iw   = inw16  + (size_t)i * 2 * DI * DM;
        const float*    cw   = conv_w + (size_t)i * DI * 4;
        const float*    cb   = conv_b + (size_t)i * DI;
        const float*    xpw  = xp_w   + (size_t)i * NXDBL * DI;
        const float*    dtw  = dtp_w  + (size_t)i * DI * DTR;
        const float*    dtb  = dtp_b  + (size_t)i * DI;
        const float*    alog = A_log  + (size_t)i * DI * DSTATE;
        const float*    dp   = Dp     + (size_t)i * DI;
        const _Float16* ow   = outw16 + (size_t)i * DM * DI;

        k_layernorm_f16<<<ROWS, DM, 0, stream>>>(hbuf, lnw, lnb, normh);

        // in_proj: (32000x128) x (128x512) -> xz, WMMA
        {
            dim3 grid((2 * DI) / 64, ROWS / 16);
            k_gemm_wmma<<<grid, 128, 0, stream>>>(normh, iw, nullptr, xz,
                                                  ROWS, 2 * DI, DM);
        }

        k_conv_silu<<<elemBlocks, 256, 0, stream>>>(xz, cw, cb, ubuf);
        k_xdbl<<<ROWS, 64, 0, stream>>>(ubuf, xpw, xdbl);
        k_dtproj<<<ROWS, DI, 0, stream>>>(xdbl, dtw, dtb, dtbuf);
        k_scan<<<BATCH, DI, 0, stream>>>(dtbuf, xdbl, ubuf, alog, xz);
        k_ymod<<<elemBlocks, 256, 0, stream>>>(xz, ubuf, dp, ymod);

        // out_proj + residual: (32000x256) x (256x128) + h -> h, WMMA
        {
            dim3 grid(DM / 64, ROWS / 16);
            k_gemm_wmma<<<grid, 128, 0, stream>>>(ymod, ow, hbuf, hbuf,
                                                  ROWS, DM, DI);
        }
    }

    k_final_ln_pool<<<BATCH, DM, 0, stream>>>(hbuf, fn_w, fn_b, pooled);
    k_head<<<BATCH, DM, 0, stream>>>(pooled, head_w, head_b, out);
}
